// Quantizer_76716705841362
// MI455X (gfx1250) — matrix operand order repaired and checked by reading, not compiled, not run on hardware
//
#include <hip/hip_runtime.h>

#define EMBED_DIM 64
#define N_CODES 1024
#define N_ROWS 65536
#define CHUNK_CODES 128
#define ROWS_PER_WG 128

typedef __attribute__((ext_vector_type(16))) __bf16 v16bf;
typedef __attribute__((ext_vector_type(8)))  float  v8f;

union BF16x16 { v16bf v; unsigned short s[16]; uint4 q[2]; };
union F8x     { v8f v; float f[8]; };

__device__ __forceinline__ unsigned short bf16_trunc(float x){
  union { float f; unsigned u; } c; c.f = x; return (unsigned short)(c.u >> 16);
}
__device__ __forceinline__ float bf16_to_f32(unsigned short s){
  union { float f; unsigned u; } c; c.u = ((unsigned)s) << 16; return c.f;
}

// LDS byte-offset of a __shared__ object (generic -> addrspace(3) -> 32-bit offset)
#define LDS_OFF(p) ((unsigned)(uintptr_t)(__attribute__((address_space(3))) const void*)(p))

// CDNA5 async memory->LDS copy (GVS mode: SGPR base + 32-bit VGPR offset), ASYNCcnt-tracked.
__device__ __forceinline__ void async_load_b128(unsigned lds_off, unsigned voff,
                                                const void* sbase){
  asm volatile("global_load_async_to_lds_b128 %0, %1, %2"
               :: "v"(lds_off), "v"(voff),
                  "s"((unsigned long long)(uintptr_t)sbase)
               : "memory");
}
__device__ __forceinline__ void wait_asynccnt0(){
  asm volatile("s_wait_asynccnt 0x0" ::: "memory");
}

// ---- prep: split codebook into bf16 hi/lo, compute ||e_k||^2 ----
__global__ void vq_prep_codes(const float* __restrict__ ew,
                              unsigned short* __restrict__ ehi,
                              unsigned short* __restrict__ elo,
                              float* __restrict__ enorm){
  int c = blockIdx.x * blockDim.x + threadIdx.x;
  if (c >= N_CODES) return;
  const float* row = ew + (size_t)c * EMBED_DIM;
  float n = 0.f;
  #pragma unroll
  for (int d = 0; d < EMBED_DIM; ++d){
    float x = row[d];
    n += x * x;
    unsigned short h = bf16_trunc(x);
    float r = x - bf16_to_f32(h);          // exact (prefix subtraction)
    ehi[(size_t)c*EMBED_DIM + d] = h;
    elo[(size_t)c*EMBED_DIM + d] = bf16_trunc(r);
  }
  enorm[c] = n;
}

__global__ void vq_init_diff(float* p){
  if (threadIdx.x == 0 && blockIdx.x == 0) *p = 0.f;
}

// ---- main: WMMA distance GEMM + argmin + gather + diff ----
__global__ __launch_bounds__(256) void vq_main(
    const float* __restrict__ z_e,
    const float* __restrict__ embed,            // f32 codes for exact gather
    const unsigned short* __restrict__ ehi,
    const unsigned short* __restrict__ elo,
    const float* __restrict__ enorm,
    float* __restrict__ out_zq,                 // [N_ROWS*64]
    float* __restrict__ out_diff,               // scalar
    float* __restrict__ out_ind)                // [N_ROWS] (float-cast indices)
{
  __shared__ unsigned short sHi[CHUNK_CODES * EMBED_DIM]; // 16 KB
  __shared__ unsigned short sLo[CHUNK_CODES * EMBED_DIM]; // 16 KB
  __shared__ float sNorm[CHUNK_CODES];
  __shared__ int   sIdx[ROWS_PER_WG];

  const int tid  = threadIdx.x;
  const int wave = tid >> 5;
  const int lane = tid & 31;
  const int half = lane >> 4;
  const int l16  = lane & 15;
  const int rowBase = blockIdx.x * ROWS_PER_WG + wave * 16;

  // ---- load this wave's 16 rows of z_e as split-bf16 A operands ----
  // A layout (16x32 bf16): lane -> row l16; halves 0-7: K = ks*32 + 8*half + h
  //                                         halves 8-15: K = ks*32 + 16 + 8*half + h
  v16bf aHi[2], aLo[2];
  {
    const float* xrow = z_e + (size_t)(rowBase + l16) * EMBED_DIM;
    #pragma unroll
    for (int ks = 0; ks < 2; ++ks){
      BF16x16 h, l;
      #pragma unroll
      for (int chunk = 0; chunk < 2; ++chunk){
        const int kb = ks*32 + chunk*16 + half*8;
        const float4* p = (const float4*)(xrow + kb);
        float4 f0 = p[0], f1 = p[1];
        float xs[8] = {f0.x,f0.y,f0.z,f0.w,f1.x,f1.y,f1.z,f1.w};
        #pragma unroll
        for (int j = 0; j < 8; ++j){
          unsigned short hb = bf16_trunc(xs[j]);
          float r = xs[j] - bf16_to_f32(hb);
          h.s[chunk*8 + j] = hb;
          l.s[chunk*8 + j] = bf16_trunc(r);
        }
      }
      aHi[ks] = h.v; aLo[ks] = l.v;
    }
  }

  float best[8];
  int   bidx[8];
  #pragma unroll
  for (int v = 0; v < 8; ++v){ best[v] = 3.4e38f; bidx[v] = 0; }

  const unsigned ldsHi = LDS_OFF(sHi);
  const unsigned ldsLo = LDS_OFF(sLo);

  for (int ch = 0; ch < N_CODES / CHUNK_CODES; ++ch){
    __syncthreads();   // previous chunk fully consumed before overwrite
    // stage split-bf16 code chunk into LDS via CDNA5 async memory->LDS path
    {
      const void* gH = (const void*)(ehi + (size_t)ch * CHUNK_CODES * EMBED_DIM);
      const void* gL = (const void*)(elo + (size_t)ch * CHUNK_CODES * EMBED_DIM);
      #pragma unroll
      for (int i = 0; i < 4; ++i){
        const unsigned off16 = (unsigned)(tid + i*256) * 16u;  // 16 KB / 256 thr / 16 B
        async_load_b128(ldsHi + off16, off16, gH);
        async_load_b128(ldsLo + off16, off16, gL);
      }
      if (tid < CHUNK_CODES) sNorm[tid] = enorm[ch * CHUNK_CODES + tid];
      wait_asynccnt0();
    }
    __syncthreads();

    #pragma unroll
    for (int t = 0; t < CHUNK_CODES / 16; ++t){
      const int lc = t*16 + l16;      // B layout: lane -> code column l16
      v8f acc = {};
      #pragma unroll
      for (int ks = 0; ks < 2; ++ks){
        // B (32x16): halves 0-7: K = ks*32 + 8*half + h; halves 8-15: K = ks*32 + 16 + 8*half + h
        BF16x16 bH, bL;
        const uint4* pH = (const uint4*)(sHi + lc*EMBED_DIM + ks*32 + half*8);
        const uint4* pL = (const uint4*)(sLo + lc*EMBED_DIM + ks*32 + half*8);
        bH.q[0] = pH[0]; bH.q[1] = pH[2];
        bL.q[0] = pL[0]; bL.q[1] = pL[2];
        // split-precision: x.e ~= xh.eh + xl.eh + xh.el  (f32 accumulate)
        acc = __builtin_amdgcn_wmma_f32_16x16x32_bf16(false, aHi[ks], false, bH.v, (short)0, acc, false, false);
        acc = __builtin_amdgcn_wmma_f32_16x16x32_bf16(false, aLo[ks], false, bH.v, (short)0, acc, false, false);
        acc = __builtin_amdgcn_wmma_f32_16x16x32_bf16(false, aHi[ks], false, bL.v, (short)0, acc, false, false);
      }
      const int   code = ch*CHUNK_CODES + t*16 + l16;
      const float en   = sNorm[t*16 + l16];
      F8x a; a.v = acc;
      #pragma unroll
      for (int v = 0; v < 8; ++v){
        float d = en - 2.0f * a.f[v];   // ||x||^2 dropped: constant per row
        if (d < best[v]) { best[v] = d; bidx[v] = code; }
      }
    }
  }

  // reduce argmin across the 16 code-columns (lanes within each half-wave)
  #pragma unroll
  for (int m = 1; m < 16; m <<= 1){
    #pragma unroll
    for (int v = 0; v < 8; ++v){
      float od = __shfl_xor(best[v], m, 32);
      int   oi = __shfl_xor(bidx[v], m, 32);
      if (od < best[v]) { best[v] = od; bidx[v] = oi; }
    }
  }
  // C layout: lane half h covers rows 8*h + v
  if (l16 == 0){
    #pragma unroll
    for (int v = 0; v < 8; ++v){
      int r = wave*16 + half*8 + v;                  // row local to WG
      sIdx[r] = bidx[v];
      out_ind[(size_t)blockIdx.x * ROWS_PER_WG + r] = (float)bidx[v];
    }
  }
  __syncthreads();

  // ---- gather z_q (exact f32 codebook), write out, accumulate diff ----
  {
    const int rloc = tid >> 1;              // 0..127
    const int dh   = (tid & 1) * 32;        // half of the 64 dims
    const size_t grow = ((size_t)blockIdx.x * ROWS_PER_WG + rloc) * EMBED_DIM;
    const int idx = sIdx[rloc];
    const float4* qrow = (const float4*)(embed + (size_t)idx * EMBED_DIM + dh);
    const float4* xrow = (const float4*)(z_e + grow + dh);
    float4* orow = (float4*)(out_zq + grow + dh);
    float s = 0.f;
    #pragma unroll
    for (int j = 0; j < 8; ++j){
      float4 q = qrow[j];
      float4 x = xrow[j];
      orow[j] = q;                           // z_q_st == z_q numerically
      float dx = q.x-x.x, dy = q.y-x.y, dz = q.z-x.z, dw = q.w-x.w;
      s += dx*dx + dy*dy + dz*dz + dw*dw;
    }
    #pragma unroll
    for (int m = 1; m < 32; m <<= 1) s += __shfl_xor(s, m, 32);
    if (lane == 0){
      const float scale = 1.25f / ((float)N_ROWS * (float)EMBED_DIM);
      atomicAdd(out_diff, s * scale);        // diff = 1.25 * mean((zq-ze)^2)
    }
  }
}

extern "C" void kernel_launch(void* const* d_in, const int* in_sizes, int n_in,
                              void* d_out, int out_size, void* d_ws, size_t ws_size,
                              hipStream_t stream) {
  const float* z_e   = (const float*)d_in[0];
  const float* embed = (const float*)d_in[1];

  float* out      = (float*)d_out;
  float* out_zq   = out;                                   // [N_ROWS*64]
  float* out_diff = out + (size_t)N_ROWS * EMBED_DIM;      // [1]
  float* out_ind  = out_diff + 1;                          // [N_ROWS]

  unsigned short* ehi = (unsigned short*)d_ws;
  unsigned short* elo = ehi + (size_t)N_CODES * EMBED_DIM;
  float* enorm        = (float*)(elo + (size_t)N_CODES * EMBED_DIM);

  vq_prep_codes<<<(N_CODES + 255) / 256, 256, 0, stream>>>(embed, ehi, elo, enorm);
  vq_init_diff<<<1, 32, 0, stream>>>(out_diff);
  vq_main<<<N_ROWS / ROWS_PER_WG, 256, 0, stream>>>(
      z_e, embed, ehi, elo, enorm, out_zq, out_diff, out_ind);
}
